// VectorQuantiser_33157147525408
// MI455X (gfx1250) — matrix operand order repaired and checked by reading, not compiled, not run on hardware
//
#include <hip/hip_runtime.h>
#include <hip/hip_bf16.h>

// ---------------------------------------------------------------------------
// MI455X / gfx1250 VQ forward.
//  - All three GEMMs run on v_wmma_f32_16x16x32_f16 (wave32, f32 accum).
//  - Distance kernel double-buffers codebook tiles in LDS using CDNA5
//    GLOBAL_LOAD_ASYNC_TO_LDS_B128 (ASYNCcnt) so the 16KB/tile copy overlaps
//    the WMMA work; zn A-operands stay resident in VGPRs the whole kernel.
//  - argmin drops the z_sq term (constant per row) -> score = c_sq - 2*dot.
//  - Loss reduced deterministically (per-row-tile partials, fixed-order sum).
// ---------------------------------------------------------------------------

typedef _Float16 half8 __attribute__((ext_vector_type(8)));
typedef _Float16 v16h  __attribute__((ext_vector_type(16)));
typedef float    v8f   __attribute__((ext_vector_type(8)));
typedef float    f4v   __attribute__((ext_vector_type(4)));

#define N_ROWS  8192
#define F_DIM   512
#define D_DIM   64
#define C_CODES 8192
#define EPSF    1e-12f
#define BETAF   0.25f

__device__ __forceinline__ v8f wmma32(v16h a, v16h b, v8f c) {
  // D = A(16x32 f16) * B(32x16 f16) + C(16x16 f32)
  return __builtin_amdgcn_wmma_f32_16x16x32_f16(false, a, false, b,
                                                (short)0, c, false, false);
}

__device__ __forceinline__ v16h mk16(half8 lo, half8 hi) {
  v16h r;
#pragma unroll
  for (int i = 0; i < 8; ++i) { r[i] = lo[i]; r[i + 8] = hi[i]; }
  return r;
}

// --- CDNA5 async global->LDS copy (ASYNCcnt tracked, no VGPR round trip) ---
typedef __attribute__((address_space(3))) void lds_void;

__device__ __forceinline__ unsigned lds_addr_of(void* p) {
#if defined(__HIP_DEVICE_COMPILE__)
  // addrspacecast generic->LDS, ptrtoint (LDS ptr is 32-bit on device)
  return (unsigned)(size_t)(lds_void*)p;
#else
  (void)p;
  return 0u;
#endif
}

__device__ __forceinline__ void async_cp_b128(unsigned lds_byte, const void* g) {
  // VDST operand = LDS byte address in a VGPR; per-lane 16B transfer.
  asm volatile("global_load_async_to_lds_b128 %0, %1, off"
               :: "v"(lds_byte), "v"(g) : "memory");
}

#define S_WAIT_ASYNCCNT(n) asm volatile("s_wait_asynccnt " #n ::: "memory")

// ---------------------------------------------------------------------------
// Kernel 1: row-normalize codebook; emit f32 + f16 copies and c_sq = sum(cn^2).
// One wave (32 lanes) per row of 64; lane handles 2 elements.
// ---------------------------------------------------------------------------
__global__ void k_norm_codebook(const float* __restrict__ cb,
                                float* __restrict__ cn,
                                _Float16* __restrict__ cnh,
                                float* __restrict__ csq) {
  int wave = threadIdx.x >> 5;
  int lane = threadIdx.x & 31;
  int row  = blockIdx.x * 8 + wave;
  const float* p = cb + (size_t)row * D_DIM + lane * 2;
  float x0 = p[0], x1 = p[1];
  float s = x0 * x0 + x1 * x1;
#pragma unroll
  for (int m = 16; m >= 1; m >>= 1) s += __shfl_xor(s, m, 32);
  float inv = rsqrtf(s * (1.0f / D_DIM) + EPSF);
  float y0 = x0 * inv, y1 = x1 * inv;
  float* q = cn + (size_t)row * D_DIM + lane * 2;
  q[0] = y0; q[1] = y1;
  _Float16* qh = cnh + (size_t)row * D_DIM + lane * 2;
  qh[0] = (_Float16)y0; qh[1] = (_Float16)y1;
  if (lane == 0) csq[row] = s * inv * inv;
}

// ---------------------------------------------------------------------------
// Kernels 2a/2b: transpose + f32->f16 convert the weights so WMMA B-operand
// loads become contiguous 32B per lane.
// ---------------------------------------------------------------------------
__global__ void k_t_win(const float* __restrict__ Win, _Float16* __restrict__ WinT) {
  int t = blockIdx.x * blockDim.x + threadIdx.x;   // 64*512 threads
  int n = t >> 9, k = t & 511;
  WinT[(size_t)n * F_DIM + k] = (_Float16)Win[(size_t)k * D_DIM + n];
}
__global__ void k_t_wout(const float* __restrict__ Wout, _Float16* __restrict__ WoutT) {
  int t = blockIdx.x * blockDim.x + threadIdx.x;   // 512*64 threads
  int n = t >> 6, k = t & 63;
  WoutT[(size_t)n * D_DIM + k] = (_Float16)Wout[(size_t)k * F_DIM + n];
}

// ---------------------------------------------------------------------------
// Kernel 3: zn = normalise(z @ W_in + b_in).  One wave per 16x64 output tile;
// A converted f32->f16 on the fly; 16 K-steps x 4 N-tiles of WMMA; RMS-norm
// done in registers with 16-lane shuffle reduction.
// ---------------------------------------------------------------------------
__global__ void k_proj_norm(const float* __restrict__ z,
                            const _Float16* __restrict__ WinT,
                            const float* __restrict__ b_in,
                            float* __restrict__ zn,
                            _Float16* __restrict__ znh) {
  int wave = threadIdx.x >> 5, lane = threadIdx.x & 31;
  int rt  = blockIdx.x * 8 + wave;       // row tile (16 rows)
  int l15 = lane & 15;
  int hi  = lane >> 4;
  int kbA  = hi ? 16 : 0;                // A-layout K base
  int kstB = hi ? 16 : 0;                // B-layout K base
  int row = rt * 16 + l15;
  const float* zrow = z + (size_t)row * F_DIM;

  v8f acc[4] = {{}, {}, {}, {}};
  for (int kk = 0; kk < F_DIM; kk += 32) {
    f4v za = *(const f4v*)(zrow + kk + kbA);
    f4v zb = *(const f4v*)(zrow + kk + kbA + 4);
    f4v zc = *(const f4v*)(zrow + kk + 8 + kbA);
    f4v zd = *(const f4v*)(zrow + kk + 8 + kbA + 4);
    v16h a;
#pragma unroll
    for (int i = 0; i < 4; ++i) {
      a[i]      = (_Float16)za[i];
      a[i + 4]  = (_Float16)zb[i];
      a[i + 8]  = (_Float16)zc[i];
      a[i + 12] = (_Float16)zd[i];
    }
#pragma unroll
    for (int nt = 0; nt < 4; ++nt) {
      const _Float16* wr = WinT + (size_t)(nt * 16 + l15) * F_DIM + kk + kstB;
      v16h b = *(const v16h*)wr;         // 32B contiguous per lane
      acc[nt] = wmma32(a, b, acc[nt]);
    }
  }

  // bias + RMS-normalize each of the 8 rows this lane touches
  float vals[4][8];
  float ssq[8];
#pragma unroll
  for (int r = 0; r < 8; ++r) ssq[r] = 0.0f;
#pragma unroll
  for (int nt = 0; nt < 4; ++nt) {
    float bb = b_in[nt * 16 + l15];
#pragma unroll
    for (int r = 0; r < 8; ++r) {
      float v = acc[nt][r] + bb;
      vals[nt][r] = v;
      ssq[r] += v * v;
    }
  }
#pragma unroll
  for (int r = 0; r < 8; ++r) {
    float s = ssq[r];
    s += __shfl_xor(s, 1, 32);
    s += __shfl_xor(s, 2, 32);
    s += __shfl_xor(s, 4, 32);
    s += __shfl_xor(s, 8, 32);           // reduce over 16 lanes (rows stay split)
    ssq[r] = rsqrtf(s * (1.0f / D_DIM) + EPSF);
  }
#pragma unroll
  for (int r = 0; r < 8; ++r) {
    int grow = rt * 16 + (hi ? 8 + r : r);
#pragma unroll
    for (int nt = 0; nt < 4; ++nt) {
      int col = nt * 16 + l15;
      float v = vals[nt][r] * ssq[r];
      zn[(size_t)grow * D_DIM + col]  = v;
      znh[(size_t)grow * D_DIM + col] = (_Float16)v;
    }
  }
}

// ---------------------------------------------------------------------------
// Kernel 4: fused distance GEMM + argmin.  8 waves/block each own a 16-row
// slab.  128-code cn tiles (16KB) are double-buffered in LDS with
// global_load_async_to_lds_b128: tile i+1 streams in while tile i is
// consumed by WMMA.  score = c_sq - 2*(zn . cn).
// ---------------------------------------------------------------------------
__global__ void k_argmin(const _Float16* __restrict__ znh,
                         const _Float16* __restrict__ cnh,
                         const float* __restrict__ csq,
                         int* __restrict__ idx,
                         float* __restrict__ idx_out) {
  __shared__ __align__(16) _Float16 lcn[2][128 * 64];   // 2 x 16KB tiles

  int t = threadIdx.x;
  int wave = t >> 5, lane = t & 31;
  int rt  = blockIdx.x * 8 + wave;
  int l15 = lane & 15, hi = lane >> 4;
  int kbA = hi ? 16 : 0, kstB = hi ? 16 : 0;
  int row = rt * 16 + l15;

  unsigned lbase0 = lds_addr_of(&lcn[0][0]) + (unsigned)t * 16u;
  unsigned lbase1 = lds_addr_of(&lcn[1][0]) + (unsigned)t * 16u;

  // prologue: kick off tile 0 (each thread owns 4 x 16B chunks)
  {
    const char* g = (const char*)cnh + (size_t)t * 16;
#pragma unroll
    for (int j = 0; j < 4; ++j)
      async_cp_b128(lbase0 + j * 4096u, g + j * 4096);
  }

  // zn A-operands (K=0..63) stay resident in VGPRs for the whole kernel
  const _Float16* zr = znh + (size_t)row * D_DIM;
  v16h a0 = mk16(*(const half8*)(zr + kbA),      *(const half8*)(zr + 8 + kbA));
  v16h a1 = mk16(*(const half8*)(zr + 32 + kbA), *(const half8*)(zr + 40 + kbA));

  float best[8]; int bidx[8];
#pragma unroll
  for (int r = 0; r < 8; ++r) { best[r] = 3.402823466e38f; bidx[r] = 0; }

  for (int it = 0; it < 64; ++it) {           // 64 tiles of 128 codes
    int buf = it & 1;
    if (it + 1 < 64) {
      // stream next tile into the other buffer (safe: last read of that
      // buffer was fenced by the barrier at the end of iteration it-1)
      const char* g = (const char*)cnh + (size_t)(it + 1) * 16384 + (size_t)t * 16;
      unsigned l = (buf ? lbase0 : lbase1);
#pragma unroll
      for (int j = 0; j < 4; ++j)
        async_cp_b128(l + j * 4096u, g + j * 4096);
      S_WAIT_ASYNCCNT(4);    // tile `it` complete (loads retire in order)
    } else {
      S_WAIT_ASYNCCNT(0);
    }
    __syncthreads();         // all 8 waves' slices of tile `it` are in LDS

#pragma unroll
    for (int st = 0; st < 8; ++st) {
      int n0 = st * 16;
      const _Float16* br = &lcn[buf][(n0 + l15) * 64];
      v16h b0 = *(const v16h*)(br + kstB);        // ds_load path
      v16h b1 = *(const v16h*)(br + 32 + kstB);
      v8f acc = {};
      acc = wmma32(a0, b0, acc);
      acc = wmma32(a1, b1, acc);
      int code = it * 128 + n0 + l15;
      float cs = csq[code];                       // L2-resident broadcast
#pragma unroll
      for (int r = 0; r < 8; ++r) {
        float d = cs - 2.0f * acc[r];
        if (d < best[r]) { best[r] = d; bidx[r] = code; }
      }
    }
    __syncthreads();         // everyone done reading buf before it refills
  }

  // reduce (min,idx) across the 16 lanes holding each row; tie -> lower idx
#pragma unroll
  for (int r = 0; r < 8; ++r) {
#pragma unroll
    for (int m = 1; m < 16; m <<= 1) {
      float ov = __shfl_xor(best[r], m, 32);
      int   oi = __shfl_xor(bidx[r], m, 32);
      if (ov < best[r] || (ov == best[r] && oi < bidx[r])) { best[r] = ov; bidx[r] = oi; }
    }
    if (l15 == 0) {
      int grow = rt * 16 + (hi ? 8 + r : r);
      idx[grow]     = bidx[r];
      idx_out[grow] = (float)bidx[r];
    }
  }
}

// ---------------------------------------------------------------------------
// Kernel 5: out = cn[idx] @ W_out + b_out (WMMA) and per-row-tile loss
// partials (f32, deterministic).  Forward loss = (1+beta)*mean((zn-codes)^2).
// ---------------------------------------------------------------------------
__global__ void k_output(const int* __restrict__ idx,
                         const _Float16* __restrict__ cnh,
                         const float* __restrict__ cn,
                         const float* __restrict__ zn,
                         const _Float16* __restrict__ WoutT,
                         const float* __restrict__ b_out,
                         float* __restrict__ out,
                         float* __restrict__ loss_part) {
  int wave = threadIdx.x >> 5, lane = threadIdx.x & 31;
  int rt  = blockIdx.x * 8 + wave;
  int l15 = lane & 15, hi = lane >> 4;
  int kbA = hi ? 16 : 0, kstB = hi ? 16 : 0;
  int row = rt * 16 + l15;
  int code = idx[row];

  const _Float16* cr = cnh + (size_t)code * D_DIM;
  v16h a0 = mk16(*(const half8*)(cr + kbA),      *(const half8*)(cr + 8 + kbA));
  v16h a1 = mk16(*(const half8*)(cr + 32 + kbA), *(const half8*)(cr + 40 + kbA));

  // loss partial: lane covers (row = rt*16+l15, dims hi*32 .. hi*32+31)
  float lp = 0.0f;
  {
    const float* zr = zn + (size_t)row * D_DIM + hi * 32;
    const float* cc = cn + (size_t)code * D_DIM + hi * 32;
#pragma unroll
    for (int j = 0; j < 32; j += 4) {
      f4v zv = *(const f4v*)(zr + j);
      f4v cv = *(const f4v*)(cc + j);
#pragma unroll
      for (int i = 0; i < 4; ++i) { float d = zv[i] - cv[i]; lp += d * d; }
    }
#pragma unroll
    for (int m = 16; m >= 1; m >>= 1) lp += __shfl_xor(lp, m, 32);
    if (lane == 0) loss_part[rt] = lp;
  }

#pragma unroll 4
  for (int nt = 0; nt < 32; ++nt) {
    int col = nt * 16 + l15;
    const _Float16* wr = WoutT + (size_t)col * D_DIM;
    v16h b0 = *(const v16h*)(wr + kstB);
    v16h b1 = *(const v16h*)(wr + 32 + kstB);
    v8f acc = {};
    acc = wmma32(a0, b0, acc);
    acc = wmma32(a1, b1, acc);
    float bo = b_out[col];
#pragma unroll
    for (int r = 0; r < 8; ++r) {
      int grow = rt * 16 + (hi ? 8 + r : r);
      out[(size_t)grow * F_DIM + col] = acc[r] + bo;
    }
  }
}

// ---------------------------------------------------------------------------
// Kernel 6: deterministic fixed-order reduction of 512 loss partials.
// ---------------------------------------------------------------------------
__global__ void k_loss_reduce(const float* __restrict__ part,
                              float* __restrict__ loss_out) {
  int lane = threadIdx.x;
  float s = 0.0f;
  for (int i = lane; i < 512; i += 32) s += part[i];
#pragma unroll
  for (int m = 16; m >= 1; m >>= 1) s += __shfl_xor(s, m, 32);
  if (lane == 0)
    *loss_out = (1.0f + BETAF) * s / ((float)N_ROWS * (float)D_DIM);
}

// ---------------------------------------------------------------------------
extern "C" void kernel_launch(void* const* d_in, const int* in_sizes, int n_in,
                              void* d_out, int out_size, void* d_ws, size_t ws_size,
                              hipStream_t stream) {
  (void)in_sizes; (void)n_in; (void)out_size; (void)ws_size;
  const float* z     = (const float*)d_in[0];   // [N,F]
  const float* Win   = (const float*)d_in[1];   // [F,D]
  const float* b_in  = (const float*)d_in[2];   // [D]
  const float* cb    = (const float*)d_in[3];   // [C,D]
  const float* Wout  = (const float*)d_in[4];   // [D,F]
  const float* b_out = (const float*)d_in[5];   // [F]

  float* out      = (float*)d_out;                       // N*F
  float* loss_out = out + (size_t)N_ROWS * F_DIM;        // 1
  float* idx_out  = loss_out + 1;                        // N (as float)

  char* ws = (char*)d_ws;
  float*    zn        = (float*)(ws + 0);                //  2 MB
  _Float16* znh       = (_Float16*)(ws + 2097152);       //  1 MB
  float*    cn        = (float*)(ws + 3145728);          //  2 MB
  _Float16* cnh       = (_Float16*)(ws + 5242880);       //  1 MB
  float*    csq       = (float*)(ws + 6291456);          // 32 KB
  int*      idx       = (int*)(ws + 6324224);            // 32 KB
  _Float16* WinT      = (_Float16*)(ws + 6356992);       // 64 KB
  _Float16* WoutT     = (_Float16*)(ws + 6422528);       // 64 KB
  float*    loss_part = (float*)(ws + 6488064);          //  2 KB

  k_norm_codebook<<<C_CODES / 8, 256, 0, stream>>>(cb, cn, cnh, csq);
  k_t_win <<<(D_DIM * F_DIM) / 256, 256, 0, stream>>>(Win, WinT);
  k_t_wout<<<(F_DIM * D_DIM) / 256, 256, 0, stream>>>(Wout, WoutT);
  k_proj_norm<<<N_ROWS / 128, 256, 0, stream>>>(z, WinT, b_in, zn, znh);
  k_argmin   <<<N_ROWS / 128, 256, 0, stream>>>(znh, cnh, csq, idx, idx_out);
  k_output   <<<N_ROWS / 128, 256, 0, stream>>>(idx, cnh, cn, zn, WoutT, b_out,
                                                out, loss_part);
  k_loss_reduce<<<1, 32, 0, stream>>>(loss_part, loss_out);
}
